// Layer_35983236006477
// MI455X (gfx1250) — matrix operand order repaired and checked by reading, not compiled, not run on hardware
//
#include <hip/hip_runtime.h>

// ---------------------------------------------------------------------------
// Types
// ---------------------------------------------------------------------------
typedef __attribute__((ext_vector_type(16))) _Float16 v16h;
typedef __attribute__((ext_vector_type(8)))  _Float16 v8h;
typedef __attribute__((ext_vector_type(8)))  float    v8f;
typedef int v4i __attribute__((vector_size(16)));

#define D_   1024
#define NH_  16
#define DH_  64
#define B_   2
#define L_   2048
#define MTOT (B_ * L_)          // 4096 rows

// Async global->LDS path (gfx1250). Guarded: falls back to plain LDS stores.
#if defined(__HIP_DEVICE_COMPILE__) && \
    __has_builtin(__builtin_amdgcn_global_load_async_to_lds_b128) && \
    __has_builtin(__builtin_amdgcn_s_wait_asynccnt)
#define ATHENA_ASYNC 1
#else
#define ATHENA_ASYNC 0
#endif

__device__ __forceinline__ void cp16(_Float16* lds, const _Float16* g) {
#if ATHENA_ASYNC
  __builtin_amdgcn_global_load_async_to_lds_b128(
      (__attribute__((address_space(1))) v4i*)g,
      (__attribute__((address_space(3))) v4i*)lds, 0, 0);
#else
  *(v8h*)lds = *(const v8h*)g;
#endif
}

__device__ __forceinline__ void stage_wait() {
#if ATHENA_ASYNC
  __builtin_amdgcn_s_wait_asynccnt(0);
#endif
  __syncthreads();
}

// Build a v16h A/B fragment from two contiguous 8-half (16-byte) chunks.
__device__ __forceinline__ v16h ld_frag(const _Float16* p0, const _Float16* p1) {
  v8h a = *(const v8h*)p0;
  v8h b = *(const v8h*)p1;
  return __builtin_shufflevector(a, b, 0,1,2,3,4,5,6,7,8,9,10,11,12,13,14,15);
}

__device__ __forceinline__ v8f wmma16(v16h a, v16h b, v8f c) {
  // (neg_a, A, neg_b, B, c_mod, C, reuse_a, reuse_b)
  return __builtin_amdgcn_wmma_f32_16x16x32_f16(false, a, false, b, (short)0, c,
                                                false, false);
}

// ---------------------------------------------------------------------------
// f32 -> f16 weight conversion
// ---------------------------------------------------------------------------
__global__ void __launch_bounds__(256) cvt_f16_kernel(const float* __restrict__ s,
                                                      _Float16* __restrict__ d,
                                                      int n) {
  int i = blockIdx.x * 256 + threadIdx.x;
  if (i < n) d[i] = (_Float16)s[i];
}

// ---------------------------------------------------------------------------
// LayerNorm: one row (D=1024) per block, output f16
// ---------------------------------------------------------------------------
__global__ void __launch_bounds__(256) ln_kernel(const float* __restrict__ x,
                                                 const float* __restrict__ g,
                                                 const float* __restrict__ b,
                                                 _Float16* __restrict__ h) {
  __shared__ float red[256];
  const int row = blockIdx.x, t = threadIdx.x;
  const float* xr = x + row * D_;
  float v0 = xr[t], v1 = xr[t + 256], v2 = xr[t + 512], v3 = xr[t + 768];
  red[t] = v0 + v1 + v2 + v3;
  __syncthreads();
  for (int o = 128; o > 0; o >>= 1) {
    if (t < o) red[t] += red[t + o];
    __syncthreads();
  }
  const float mu = red[0] * (1.0f / 1024.0f);
  __syncthreads();
  float d0 = v0 - mu, d1 = v1 - mu, d2 = v2 - mu, d3 = v3 - mu;
  red[t] = d0 * d0 + d1 * d1 + d2 * d2 + d3 * d3;
  __syncthreads();
  for (int o = 128; o > 0; o >>= 1) {
    if (t < o) red[t] += red[t + o];
    __syncthreads();
  }
  const float rstd = rsqrtf(red[0] * (1.0f / 1024.0f) + 1e-5f);
  _Float16* hr = h + row * D_;
  hr[t]       = (_Float16)(d0 * rstd * g[t]       + b[t]);
  hr[t + 256] = (_Float16)(d1 * rstd * g[t + 256] + b[t + 256]);
  hr[t + 512] = (_Float16)(d2 * rstd * g[t + 512] + b[t + 512]);
  hr[t + 768] = (_Float16)(d3 * rstd * g[t + 768] + b[t + 768]);
}

// ---------------------------------------------------------------------------
// QKV GEMM: C[m,n] = sum_k A[m,k] * W[n,k] + bias[n]; scatter to q/k/v
// A: [4096][1024] f16   W: [3072][1024] f16      q/k/v: [B, NH, L, DH] f16
// Block: 256 thr (8 waves), tile 128(M) x 128(N), K step 32, double-buffered
// async global->LDS staging (next tile in flight during WMMA).
// ---------------------------------------------------------------------------
__global__ void __launch_bounds__(256) gemm_qkv_kernel(
    const _Float16* __restrict__ A, const _Float16* __restrict__ W,
    const float* __restrict__ bias, _Float16* __restrict__ qO,
    _Float16* __restrict__ kO, _Float16* __restrict__ vO) {
  __shared__ _Float16 As[2][128][32];
  __shared__ _Float16 Bs[2][128][32];
  const int t = threadIdx.x;
  const int lane = t & 31, wave = t >> 5;
  const int lr = lane & 15, lh = lane >> 4;
  const int wm = wave >> 1, wn = wave & 1;   // 4 x 2 wave grid
  const int m0 = blockIdx.y * 128, n0 = blockIdx.x * 128;
  const int rr = t >> 2, c8 = (t & 3) * 8;

  auto stage = [&](int buf, int k0) {
    cp16(&As[buf][rr][c8],      A + (m0 + rr) * D_ + k0 + c8);
    cp16(&As[buf][rr + 64][c8], A + (m0 + rr + 64) * D_ + k0 + c8);
    cp16(&Bs[buf][rr][c8],      W + (n0 + rr) * D_ + k0 + c8);
    cp16(&Bs[buf][rr + 64][c8], W + (n0 + rr + 64) * D_ + k0 + c8);
  };

  const v8f z8 = {0.f, 0.f, 0.f, 0.f, 0.f, 0.f, 0.f, 0.f};
  v8f acc[2][4];
#pragma unroll
  for (int i = 0; i < 2; ++i)
#pragma unroll
    for (int j = 0; j < 4; ++j) acc[i][j] = z8;

  stage(0, 0);
  for (int k0 = 0; k0 < D_; k0 += 32) {
    const int cur = (k0 >> 5) & 1;
    const bool has_next = (k0 + 32 < D_);
    if (has_next) {
      stage(cur ^ 1, k0 + 32);
      if (k0 + 64 < D_) {
        __builtin_prefetch(A + (m0 + rr) * D_ + k0 + 64 + c8, 0, 1);
        __builtin_prefetch(W + (n0 + rr) * D_ + k0 + 64 + c8, 0, 1);
      }
    }
#if ATHENA_ASYNC
    if (has_next) __builtin_amdgcn_s_wait_asynccnt(4);  // keep next tile in flight
    else          __builtin_amdgcn_s_wait_asynccnt(0);
#endif
    __syncthreads();

    v16h af[2], bf[4];
#pragma unroll
    for (int i = 0; i < 2; ++i) {
      const _Float16* p = &As[cur][wm * 32 + i * 16 + lr][lh * 8];
      af[i] = ld_frag(p, p + 16);
    }
#pragma unroll
    for (int j = 0; j < 4; ++j) {
      const _Float16* p = &Bs[cur][wn * 64 + j * 16 + lr][lh * 8];
      bf[j] = ld_frag(p, p + 16);
    }
#pragma unroll
    for (int i = 0; i < 2; ++i)
#pragma unroll
      for (int j = 0; j < 4; ++j) acc[i][j] = wmma16(af[i], bf[j], acc[i][j]);
    __syncthreads();
  }

#pragma unroll
  for (int i = 0; i < 2; ++i)
#pragma unroll
    for (int j = 0; j < 4; ++j)
#pragma unroll
      for (int r = 0; r < 8; ++r) {
        const int m = m0 + wm * 32 + i * 16 + r + 8 * lh;
        const int n = n0 + wn * 64 + j * 16 + lr;
        const float val = acc[i][j][r] + bias[n];
        const int which = n >> 10;
        const int hc = n & 1023;
        const int head = hc >> 6, c = hc & 63;
        const int bb = m >> 11, l = m & 2047;
        _Float16* dst = (which == 0) ? qO : ((which == 1) ? kO : vO);
        dst[(((bb * NH_ + head) * L_) + l) * DH_ + c] = (_Float16)val;
      }
}

// ---------------------------------------------------------------------------
// Fused causal flash attention. 1 block = 4 waves = 64 queries of one (b,h).
// Q pre-scaled by 1/sqrt(D)=1/32. Output -> hbuf [b, l, head*64+c] f16.
// ---------------------------------------------------------------------------
__global__ void __launch_bounds__(128) attn_kernel(const _Float16* __restrict__ Qb,
                                                   const _Float16* __restrict__ Kb,
                                                   const _Float16* __restrict__ Vb,
                                                   _Float16* __restrict__ Hb) {
  __shared__ _Float16 Ks[32][64];     // 32 keys x 64 dh
  __shared__ _Float16 Vts[64][40];    // transposed V, padded stride (16B aligned)
  __shared__ _Float16 Ps[4][16][32];  // per-wave P staging

  const int t = threadIdx.x;
  const int lane = t & 31, wave = t >> 5;
  const int lr = lane & 15, lh = lane >> 4;
  const int qt0 = blockIdx.x * 64;
  const int head = blockIdx.y, bb = blockIdx.z;
  const int qbase = qt0 + wave * 16;
  const int bh = bb * NH_ + head;
  const _Float16* Q = Qb + bh * (L_ * DH_);
  const _Float16* K = Kb + bh * (L_ * DH_);
  const _Float16* V = Vb + bh * (L_ * DH_);

  const v8f z8 = {0.f, 0.f, 0.f, 0.f, 0.f, 0.f, 0.f, 0.f};

  // Q fragments (16 queries x dh=64 => 2 A-frags), pre-scaled by 1/32
  v16h qf[2];
  {
    const _Float16* p = Q + (qbase + lr) * DH_ + lh * 8;
    qf[0] = ld_frag(p, p + 16);
    qf[1] = ld_frag(p + 32, p + 48);
    const _Float16 sc = (_Float16)0.03125f;
#pragma unroll
    for (int j = 0; j < 16; ++j) {
      qf[0][j] = qf[0][j] * sc;
      qf[1][j] = qf[1][j] * sc;
    }
  }

  v8f o[4];
#pragma unroll
  for (int nt = 0; nt < 4; ++nt) o[nt] = z8;
  float mrow[8], srow[8];
#pragma unroll
  for (int r = 0; r < 8; ++r) { mrow[r] = -3.0e38f; srow[r] = 0.0f; }

  const int kend = qt0 + 64;
  for (int kk = 0; kk < kend; kk += 32) {
    // stage K tile + transposed V tile (128 threads, 2 chunks each)
    for (int cid = t; cid < 256; cid += 128) {
      const int r = cid >> 3, c8 = (cid & 7) * 8;
      cp16(&Ks[r][c8], K + (kk + r) * DH_ + c8);
      v8h vv = *(const v8h*)(V + (kk + r) * DH_ + c8);
#pragma unroll
      for (int j = 0; j < 8; ++j) Vts[c8 + j][r] = vv[j];
    }
    stage_wait();

    if (kk <= qbase + 15) {  // uniform per wave: EXEC stays all-ones
      // S = Q * K^T  (16 queries x 32 keys)
      v8f sfr[2];
#pragma unroll
      for (int nh = 0; nh < 2; ++nh) {
        const _Float16* p0 = &Ks[nh * 16 + lr][lh * 8];
        v16h kf0 = ld_frag(p0, p0 + 16);           // dh 0..31
        v16h kf1 = ld_frag(p0 + 32, p0 + 48);      // dh 32..63
        v8f s = z8;
        s = wmma16(qf[0], kf0, s);
        s = wmma16(qf[1], kf1, s);
        sfr[nh] = s;
      }

      // causal mask + online softmax
#pragma unroll
      for (int r = 0; r < 8; ++r) {
        const int q = qbase + r + 8 * lh;
        float s0 = sfr[0][r];
        float s1 = sfr[1][r];
        s0 = (kk + lr > q) ? -1.0e30f : s0;
        s1 = (kk + 16 + lr > q) ? -1.0e30f : s1;
        float mx = fmaxf(s0, s1);
#pragma unroll
        for (int off = 1; off < 16; off <<= 1) mx = fmaxf(mx, __shfl_xor(mx, off, 32));
        const float mnew = fmaxf(mrow[r], mx);
        const float alpha = __expf(mrow[r] - mnew);
        const float p0 = __expf(s0 - mnew);
        const float p1 = __expf(s1 - mnew);
        float ps = p0 + p1;
#pragma unroll
        for (int off = 1; off < 16; off <<= 1) ps += __shfl_xor(ps, off, 32);
        srow[r] = srow[r] * alpha + ps;
        mrow[r] = mnew;
#pragma unroll
        for (int nt = 0; nt < 4; ++nt) o[nt][r] = o[nt][r] * alpha;
        Ps[wave][r + 8 * lh][lr] = (_Float16)p0;
        Ps[wave][r + 8 * lh][16 + lr] = (_Float16)p1;
      }

      // O += P * V  (A-frag from Ps, B-frag from transposed V)
      {
        const _Float16* pp = &Ps[wave][lr][lh * 8];
        v16h pa = ld_frag(pp, pp + 16);
#pragma unroll
        for (int nt = 0; nt < 4; ++nt) {
          const _Float16* vp = &Vts[nt * 16 + lr][lh * 8];
          v16h vf = ld_frag(vp, vp + 16);
          o[nt] = wmma16(pa, vf, o[nt]);
        }
      }
    }
    __syncthreads();
  }

  // normalize + store heads in [b, l, head*64+c] layout
#pragma unroll
  for (int r = 0; r < 8; ++r) {
    const float inv = (srow[r] > 0.0f) ? (1.0f / srow[r]) : 0.0f;
    const int l = qbase + r + 8 * lh;
#pragma unroll
    for (int nt = 0; nt < 4; ++nt) {
      const float val = o[nt][r] * inv;
      Hb[(bb * L_ + l) * D_ + head * DH_ + nt * 16 + lr] = (_Float16)val;
    }
  }
}

// ---------------------------------------------------------------------------
// Output GEMM + residual + exact GELU (double-buffered like gemm_qkv).
// A: heads [4096][1024] f16   W: wo [1024][1024] f16   out fp32
// ---------------------------------------------------------------------------
__global__ void __launch_bounds__(256) gemm_out_kernel(
    const _Float16* __restrict__ A, const _Float16* __restrict__ W,
    const float* __restrict__ bias, const float* __restrict__ X,
    float* __restrict__ out) {
  __shared__ _Float16 As[2][128][32];
  __shared__ _Float16 Bs[2][128][32];
  const int t = threadIdx.x;
  const int lane = t & 31, wave = t >> 5;
  const int lr = lane & 15, lh = lane >> 4;
  const int wm = wave >> 1, wn = wave & 1;
  const int m0 = blockIdx.y * 128, n0 = blockIdx.x * 128;
  const int rr = t >> 2, c8 = (t & 3) * 8;

  auto stage = [&](int buf, int k0) {
    cp16(&As[buf][rr][c8],      A + (m0 + rr) * D_ + k0 + c8);
    cp16(&As[buf][rr + 64][c8], A + (m0 + rr + 64) * D_ + k0 + c8);
    cp16(&Bs[buf][rr][c8],      W + (n0 + rr) * D_ + k0 + c8);
    cp16(&Bs[buf][rr + 64][c8], W + (n0 + rr + 64) * D_ + k0 + c8);
  };

  const v8f z8 = {0.f, 0.f, 0.f, 0.f, 0.f, 0.f, 0.f, 0.f};
  v8f acc[2][4];
#pragma unroll
  for (int i = 0; i < 2; ++i)
#pragma unroll
    for (int j = 0; j < 4; ++j) acc[i][j] = z8;

  stage(0, 0);
  for (int k0 = 0; k0 < D_; k0 += 32) {
    const int cur = (k0 >> 5) & 1;
    const bool has_next = (k0 + 32 < D_);
    if (has_next) {
      stage(cur ^ 1, k0 + 32);
      if (k0 + 64 < D_) {
        __builtin_prefetch(A + (m0 + rr) * D_ + k0 + 64 + c8, 0, 1);
        __builtin_prefetch(W + (n0 + rr) * D_ + k0 + 64 + c8, 0, 1);
      }
    }
#if ATHENA_ASYNC
    if (has_next) __builtin_amdgcn_s_wait_asynccnt(4);
    else          __builtin_amdgcn_s_wait_asynccnt(0);
#endif
    __syncthreads();

    v16h af[2], bf[4];
#pragma unroll
    for (int i = 0; i < 2; ++i) {
      const _Float16* p = &As[cur][wm * 32 + i * 16 + lr][lh * 8];
      af[i] = ld_frag(p, p + 16);
    }
#pragma unroll
    for (int j = 0; j < 4; ++j) {
      const _Float16* p = &Bs[cur][wn * 64 + j * 16 + lr][lh * 8];
      bf[j] = ld_frag(p, p + 16);
    }
#pragma unroll
    for (int i = 0; i < 2; ++i)
#pragma unroll
      for (int j = 0; j < 4; ++j) acc[i][j] = wmma16(af[i], bf[j], acc[i][j]);
    __syncthreads();
  }

#pragma unroll
  for (int i = 0; i < 2; ++i)
#pragma unroll
    for (int j = 0; j < 4; ++j)
#pragma unroll
      for (int r = 0; r < 8; ++r) {
        const int m = m0 + wm * 32 + i * 16 + r + 8 * lh;
        const int n = n0 + wn * 64 + j * 16 + lr;
        const float y = X[m * D_ + n] + acc[i][j][r] + bias[n];
        out[m * D_ + n] = 0.5f * y * (1.0f + erff(y * 0.70710678118654752f));
      }
}

// ---------------------------------------------------------------------------
// Host launcher
// ---------------------------------------------------------------------------
extern "C" void kernel_launch(void* const* d_in, const int* in_sizes, int n_in,
                              void* d_out, int out_size, void* d_ws,
                              size_t ws_size, hipStream_t stream) {
  (void)in_sizes; (void)n_in; (void)ws_size;
  const float* x    = (const float*)d_in[0];
  const float* mask = (const float*)d_in[1];
  const float* wx_w = (const float*)d_in[2];
  const float* wx_b = (const float*)d_in[3];
  const float* wo_w = (const float*)d_in[4];
  const float* wo_b = (const float*)d_in[5];
  const float* ln_g = (const float*)d_in[6];
  const float* ln_b = (const float*)d_in[7];
  float* out = (float*)d_out;

  // f16 workspace layout
  _Float16* ws   = (_Float16*)d_ws;
  _Float16* hf   = ws;                       // 4096*1024
  _Float16* wxwf = hf + MTOT * D_;           // 3072*1024
  _Float16* wowf = wxwf + 3 * D_ * D_;       // 1024*1024
  _Float16* qb   = wowf + D_ * D_;           // B*NH*L*DH
  _Float16* kb   = qb + B_ * NH_ * L_ * DH_;
  _Float16* vb   = kb + B_ * NH_ * L_ * DH_;
  _Float16* hb   = vb + B_ * NH_ * L_ * DH_; // heads, [b,l,d]

  cvt_f16_kernel<<<(3 * D_ * D_ + 255) / 256, 256, 0, stream>>>(wx_w, wxwf, 3 * D_ * D_);
  cvt_f16_kernel<<<(D_ * D_ + 255) / 256, 256, 0, stream>>>(wo_w, wowf, D_ * D_);
  ln_kernel<<<MTOT, 256, 0, stream>>>(x, ln_g, ln_b, hf);

  gemm_qkv_kernel<<<dim3(3 * D_ / 128, MTOT / 128), 256, 0, stream>>>(
      hf, wxwf, wx_b, qb, kb, vb);

  attn_kernel<<<dim3(L_ / 64, NH_, B_), 128, 0, stream>>>(qb, kb, vb, hb);

  gemm_out_kernel<<<dim3(D_ / 128, MTOT / 128), 256, 0, stream>>>(
      hb, wowf, wo_b, x, out);

  // second tuple element: the mask, passed through
  if (out_size >= 2 * MTOT * D_) {
    (void)hipMemcpyAsync(out + MTOT * D_, mask, (size_t)L_ * L_ * sizeof(float),
                         hipMemcpyDeviceToDevice, stream);
  }
}
